// BClassifier_65481071411055
// MI455X (gfx1250) — matrix operand order repaired and checked by reading, not compiled, not run on hardware
//
#include <hip/hip_runtime.h>

#define TM   128            // rows per block in main kernel
#define XSTR 520            // bf16 elems per LDS row (512 + 8 pad -> conflict-free frag reads)
#define QSTR 132            // fp32 Q row stride in LDS
#define NBIN 2048

typedef __attribute__((ext_vector_type(8)))  __bf16 bf8_t;
typedef __attribute__((ext_vector_type(16))) __bf16 bf16x16;
typedef __attribute__((ext_vector_type(8)))  float  v8f;

__device__ __forceinline__ unsigned short f2bf(float f) {
  unsigned u = __float_as_uint(f);
  u += 0x7FFFu + ((u >> 16) & 1u);      // round-to-nearest-even
  return (unsigned short)(u >> 16);
}

__device__ __forceinline__ v8f wmma_bf16(bf16x16 a, bf16x16 b, v8f c) {
  return __builtin_amdgcn_wmma_f32_16x16x32_bf16(false, a, false, b, (short)0, c, false, false);
}

__device__ __forceinline__ unsigned mapf(float v) {   // order-preserving float->uint
  unsigned u = __float_as_uint(v);
  return (u & 0x80000000u) ? ~u : (u | 0x80000000u);
}

// ---------------- small helper kernels ----------------

__global__ void init_ws_kernel(unsigned* w, int n) {
  for (int i = blockIdx.x * blockDim.x + threadIdx.x; i < n; i += gridDim.x * blockDim.x)
    w[i] = 0u;
}

__global__ void conv_bf16_kernel(const float* wl, const float* wv, const float* wq,
                                 unsigned short* wlb, unsigned short* wvb, unsigned short* wqb) {
  const int NL = 512 * 512, NV = 512 * 512, NQ = 128 * 512;
  for (int i = blockIdx.x * blockDim.x + threadIdx.x; i < NL + NV + NQ;
       i += gridDim.x * blockDim.x) {
    if (i < NL)            wlb[i] = f2bf(wl[i]);
    else if (i < NL + NV)  wvb[i - NL] = f2bf(wv[i - NL]);
    else                   wqb[i - NL - NV] = f2bf(wq[i - NL - NV]);
  }
}

__global__ void hist1_kernel(const float* preds, unsigned* hist, int N) {
  __shared__ unsigned h[NBIN];
  for (int i = threadIdx.x; i < NBIN; i += blockDim.x) h[i] = 0u;
  __syncthreads();
  for (int i = blockIdx.x * blockDim.x + threadIdx.x; i < N; i += gridDim.x * blockDim.x) {
    float p = preds[i];
    int b = (int)((p + 8.0f) * 128.0f);
    b = min(max(b, 0), NBIN - 1);
    atomicAdd(&h[b], 1u);
  }
  __syncthreads();
  for (int i = threadIdx.x; i < NBIN; i += blockDim.x)
    if (h[i]) atomicAdd(&hist[i], h[i]);
}

// scal layout (floats/uints at same base): [0]=t  [1]=gmax_bits(u) [2]=S  [3]=bin_lo  [4]=count_below(u)
__global__ void scan1_kernel(const unsigned* hist, float* scal, unsigned* scalu, int k) {
  if (threadIdx.x == 0) {
    unsigned cum = 0; int bin = NBIN - 1;
    for (int b = 0; b < NBIN; ++b) {
      unsigned c = hist[b];
      if (cum + c >= (unsigned)k) { bin = b; break; }
      cum += c;
    }
    scal[3]  = -8.0f + (float)bin * (1.0f / 128.0f);
    scalu[4] = cum;
  }
}

__global__ void hist2_kernel(const float* preds, unsigned* hist, const float* scal, int N) {
  __shared__ unsigned h[NBIN];
  for (int i = threadIdx.x; i < NBIN; i += blockDim.x) h[i] = 0u;
  __syncthreads();
  float lo = scal[3];
  const float wrec = 128.0f * 2048.0f;
  for (int i = blockIdx.x * blockDim.x + threadIdx.x; i < N; i += gridDim.x * blockDim.x) {
    float p = preds[i];
    if (p >= lo && p < lo + (1.0f / 128.0f)) {
      int b = (int)((p - lo) * wrec);
      b = min(max(b, 0), NBIN - 1);
      atomicAdd(&h[b], 1u);
    }
  }
  __syncthreads();
  for (int i = threadIdx.x; i < NBIN; i += blockDim.x)
    if (h[i]) atomicAdd(&hist[i], h[i]);
}

__global__ void scan2_kernel(const unsigned* hist, float* scal, unsigned* scalu, int k) {
  if (threadIdx.x == 0) {
    float lo = scal[3];
    unsigned cum = scalu[4];
    float t = lo + (1.0f / 128.0f);
    for (int b = 0; b < NBIN; ++b) {
      cum += hist[b];
      if (cum >= (unsigned)k) { t = lo + (float)(b + 1) * (1.0f / (128.0f * 2048.0f)); break; }
    }
    scal[0] = t;
  }
}

__global__ __launch_bounds__(256) void maxred_kernel(const float* preds, const float* scal,
                                                     unsigned* scalu, int N) {
  __shared__ float red[256];
  float t = scal[0];
  float lm = 0.0f;  // global max of p' is >= 0 since k>=1 values get zeroed
  for (int i = blockIdx.x * blockDim.x + threadIdx.x; i < N; i += gridDim.x * blockDim.x) {
    float p = preds[i];
    if (p < t) p = 0.0f;
    lm = fmaxf(lm, p);
  }
  red[threadIdx.x] = lm;
  __syncthreads();
  for (int s = 128; s > 0; s >>= 1) {
    if (threadIdx.x < s) red[threadIdx.x] = fmaxf(red[threadIdx.x], red[threadIdx.x + s]);
    __syncthreads();
  }
  if (threadIdx.x == 0) atomicMax(&scalu[1], __float_as_uint(fmaxf(red[0], 0.0f)));
}

__global__ __launch_bounds__(256) void expsum_kernel(const float* preds, float* scal,
                                                     const unsigned* scalu, int N) {
  __shared__ float red[256];
  float t = scal[0];
  float m = __uint_as_float(scalu[1]);
  float s = 0.0f;
  for (int i = blockIdx.x * blockDim.x + threadIdx.x; i < N; i += gridDim.x * blockDim.x) {
    float p = preds[i];
    if (p < t) p = 0.0f;
    s += expf(p - m);
  }
  red[threadIdx.x] = s;
  __syncthreads();
  for (int st = 128; st > 0; st >>= 1) {
    if (threadIdx.x < st) red[threadIdx.x] += red[threadIdx.x + st];
    __syncthreads();
  }
  if (threadIdx.x == 0) atomicAdd(&scal[2], red[0]);
}

__global__ __launch_bounds__(256) void argmaxc_kernel(const float* c, unsigned long long* keys,
                                                      int N) {
  __shared__ unsigned long long kk[512];
  unsigned long long b0 = 0ull, b1 = 0ull;
  for (int i = blockIdx.x * blockDim.x + threadIdx.x; i < N; i += gridDim.x * blockDim.x) {
    float v0 = c[i * 2 + 0], v1 = c[i * 2 + 1];
    unsigned long long k0 = ((unsigned long long)mapf(v0) << 32) |
                            (unsigned long long)(0xFFFFFFFFu - (unsigned)i);
    unsigned long long k1 = ((unsigned long long)mapf(v1) << 32) |
                            (unsigned long long)(0xFFFFFFFFu - (unsigned)i);
    if (k0 > b0) b0 = k0;
    if (k1 > b1) b1 = k1;
  }
  kk[threadIdx.x * 2 + 0] = b0;
  kk[threadIdx.x * 2 + 1] = b1;
  __syncthreads();
  for (int s = 128; s > 0; s >>= 1) {
    if (threadIdx.x < s) {
      if (kk[(threadIdx.x + s) * 2 + 0] > kk[threadIdx.x * 2 + 0])
        kk[threadIdx.x * 2 + 0] = kk[(threadIdx.x + s) * 2 + 0];
      if (kk[(threadIdx.x + s) * 2 + 1] > kk[threadIdx.x * 2 + 1])
        kk[threadIdx.x * 2 + 1] = kk[(threadIdx.x + s) * 2 + 1];
    }
    __syncthreads();
  }
  if (threadIdx.x == 0) {
    atomicMax(&keys[0], kk[0]);
    atomicMax(&keys[1], kk[1]);
  }
}

// exact fp32 q_max for the 2 argmax rows
__global__ __launch_bounds__(256) void mfeat_kernel(const float* feats, const float* preds,
                                                    const float* W_lin, const float* b_lin,
                                                    const float* W_q, const float* b_q,
                                                    const unsigned long long* keys,
                                                    const float* scal, const unsigned* scalu,
                                                    float* qmax, int N) {
  __shared__ float xb[512];
  __shared__ float fb[512];
  int cls = blockIdx.x;
  unsigned mi = 0xFFFFFFFFu - (unsigned)(keys[cls] & 0xFFFFFFFFull);
  if (mi >= (unsigned)N) mi = 0;
  float t = scal[0], m = __uint_as_float(scalu[1]), S = scal[2];
  float p = preds[mi];
  if (p < t) p = 0.0f;
  float mult = 1.0f + expf(p - m) / S;
  for (int d = threadIdx.x; d < 512; d += 256)
    xb[d] = feats[(size_t)mi * 512 + d] * mult;
  __syncthreads();
  for (int n = threadIdx.x; n < 512; n += 256) {
    float a = b_lin[n];
    const float* w = W_lin + (size_t)n * 512;
    for (int d = 0; d < 512; ++d) a += xb[d] * w[d];
    fb[n] = fmaxf(a, 0.0f);
  }
  __syncthreads();
  for (int q = threadIdx.x; q < 128; q += 256) {
    float a = b_q[q];
    const float* w = W_q + (size_t)q * 512;
    for (int d = 0; d < 512; ++d) a += fb[d] * w[d];
    qmax[cls * 128 + q] = tanhf(a);
  }
}

// ---------------- main fused WMMA kernel ----------------
// per block: 128 rows. stage1 f=relu(X Wlin^T), stage2 Q=tanh(f Wq^T) -> e=exp(Q qmax^T/sqrt(128)),
// stage3 V=f Wv^T folded into B partials. f/V/Q never leave the WGP.
// A/K fragments for a wave's 16-row slab are hoisted into 128 VGPRs and reused
// across all n-tiles, so the inner loop is pure global_load(W) + v_wmma.
__global__ __launch_bounds__(256, 1)
void main_kernel(const float* __restrict__ feats, const float* __restrict__ preds,
                 const unsigned short* __restrict__ WlinB, const unsigned short* __restrict__ WvB,
                 const unsigned short* __restrict__ WqB, const float* __restrict__ b_lin,
                 const float* __restrict__ b_v, const float* __restrict__ b_q,
                 const float* __restrict__ scal, const unsigned* __restrict__ scalu,
                 const float* __restrict__ qmax, float* __restrict__ Aout,
                 float* __restrict__ esum_g, float* __restrict__ BaccG, int N) {
  extern __shared__ char smem[];
  unsigned short* XsU = (unsigned short*)smem;            // TM*XSTR bf16
  unsigned short* FsU = XsU + TM * XSTR;                  // TM*XSTR bf16
  float* fbase = (float*)(FsU + TM * XSTR);
  float* Es   = fbase;                                    // [TM][2]
  float* Bacc = fbase + 256;                              // [2][512]
  float* esl  = fbase + 1280;                             // [2]
  float* mult = fbase + 1282;                             // [TM]
  float* Qs   = (float*)XsU;                              // reuse X region after stage1

  const int tid  = threadIdx.x;
  const int lane = tid & 31;
  const int wave = tid >> 5;
  const int l15  = lane & 15;
  const int hi   = lane >> 4;          // 0 for lanes 0-15, 1 for lanes 16-31
  const int m0   = wave * 16;
  const long long g0 = (long long)blockIdx.x * TM;

  // zero partials + per-row multiplier
  for (int i = tid; i < 1026; i += 256) {
    if (i < 1024) Bacc[i] = 0.0f; else esl[i - 1024] = 0.0f;
  }
  {
    float t = scal[0], m = __uint_as_float(scalu[1]), S = scal[2];
    for (int r = tid; r < TM; r += 256) {
      long long gr = g0 + r;
      float ml = 1.0f;
      if (gr < (long long)N) {
        float p = preds[gr];
        if (p < t) p = 0.0f;
        ml = 1.0f + expf(p - m) / S;
      }
      mult[r] = ml;
    }
  }
  __syncthreads();

  // stage X load: feats*(1+score) -> bf16 LDS
  for (int idx = tid; idx < TM * 128; idx += 256) {   // 128 float4-groups per row
    int r = idx >> 7, cg = idx & 127;
    long long gr = g0 + r;
    float4 v = make_float4(0.f, 0.f, 0.f, 0.f);
    if (gr < (long long)N) v = *(const float4*)(feats + gr * 512 + cg * 4);
    float ml = mult[r];
    unsigned p0 = (unsigned)f2bf(v.x * ml) | ((unsigned)f2bf(v.y * ml) << 16);
    unsigned p1 = (unsigned)f2bf(v.z * ml) | ((unsigned)f2bf(v.w * ml) << 16);
    *(uint2*)(XsU + r * XSTR + cg * 4) = make_uint2(p0, p1);
  }
  __syncthreads();

  // ---- stage 1: f = relu(X Wlin^T + b_lin), bf16 into Fs ----
  {
    const __bf16* arow = (const __bf16*)(XsU + (m0 + l15) * XSTR);
    bf16x16 af[16];                              // hoisted A fragments (128 VGPRs)
#pragma unroll
    for (int kt = 0; kt < 16; ++kt) {
      int k0 = (kt << 5) + (hi << 4);
      bf8_t a0 = *(const bf8_t*)(arow + k0);
      bf8_t a1 = *(const bf8_t*)(arow + k0 + 8);
      af[kt] = __builtin_shufflevector(a0, a1, 0, 1, 2, 3, 4, 5, 6, 7,
                                       8, 9, 10, 11, 12, 13, 14, 15);
    }
    const __bf16* wbase = (const __bf16*)WlinB + (hi << 4);
    for (int nt = 0; nt < 32; ++nt) {
      int n = nt * 16 + l15;
      const __bf16* wrow = wbase + n * 512;
      if (nt + 1 < 32)
        __builtin_prefetch((const void*)(wbase + (n + 16) * 512), 0, 3);  // global_prefetch_b8
      v8f acc = {};
#pragma unroll
      for (int kt = 0; kt < 16; ++kt) {
        bf16x16 bv = *(const bf16x16*)(wrow + (kt << 5));
        acc = wmma_bf16(af[kt], bv, acc);
      }
      float bias = b_lin[n];
#pragma unroll
      for (int v = 0; v < 8; ++v) {
        float fv = fmaxf(acc[v] + bias, 0.0f);
        FsU[(m0 + (hi << 3) + v) * XSTR + n] = f2bf(fv);
      }
    }
  }
  __syncthreads();

  // ---- load F fragments once; shared by stage 2a (Q) and stage 3 (V) ----
  bf16x16 ff[16];
  {
    const __bf16* frow = (const __bf16*)(FsU + (m0 + l15) * XSTR);
#pragma unroll
    for (int kt = 0; kt < 16; ++kt) {
      int k0 = (kt << 5) + (hi << 4);
      bf8_t a0 = *(const bf8_t*)(frow + k0);
      bf8_t a1 = *(const bf8_t*)(frow + k0 + 8);
      ff[kt] = __builtin_shufflevector(a0, a1, 0, 1, 2, 3, 4, 5, 6, 7,
                                       8, 9, 10, 11, 12, 13, 14, 15);
    }
  }

  // ---- stage 2a: Q = tanh(f Wq^T + b_q) -> Qs (overwrites X region) ----
  {
    const __bf16* wbase = (const __bf16*)WqB + (hi << 4);
    for (int nt = 0; nt < 8; ++nt) {
      int n = nt * 16 + l15;
      const __bf16* wrow = wbase + n * 512;
      if (nt + 1 < 8)
        __builtin_prefetch((const void*)(wbase + (n + 16) * 512), 0, 3);
      v8f acc = {};
#pragma unroll
      for (int kt = 0; kt < 16; ++kt) {
        bf16x16 bv = *(const bf16x16*)(wrow + (kt << 5));
        acc = wmma_bf16(ff[kt], bv, acc);
      }
      float bias = b_q[n];
#pragma unroll
      for (int v = 0; v < 8; ++v)
        Qs[(m0 + (hi << 3) + v) * QSTR + n] = tanhf(acc[v] + bias);
    }
  }
  __syncthreads();

  // ---- stage 2b: e = exp(Q . qmax / sqrt(128)) (unnormalized softmax numerators) ----
  {
    int r = tid >> 1, cc = tid & 1;
    const float* qrow = Qs + r * QSTR;
    const float* qm = qmax + cc * 128;
    float dot = 0.0f;
#pragma unroll 8
    for (int q = 0; q < 128; ++q) dot += qrow[q] * qm[q];
    long long gr = g0 + r;
    float e = 0.0f;
    if (gr < (long long)N) e = expf(dot * 0.0883883476483184f);  // 1/sqrt(128)
    Es[r * 2 + cc] = e;
    atomicAdd(&esl[cc], e);
    if (gr < (long long)N) Aout[gr * 2 + cc] = e;
  }
  __syncthreads();
  if (tid < 2) atomicAdd(&esum_g[tid], esl[tid]);

  // ---- stage 3: V = f Wv^T + b_v, folded into B partials ----
  {
    float e0[8], e1[8];
#pragma unroll
    for (int v = 0; v < 8; ++v) {
      int m = m0 + (hi << 3) + v;
      e0[v] = Es[m * 2 + 0];
      e1[v] = Es[m * 2 + 1];
    }
    const __bf16* wbase = (const __bf16*)WvB + (hi << 4);
    for (int nt = 0; nt < 32; ++nt) {
      int n = nt * 16 + l15;
      const __bf16* wrow = wbase + n * 512;
      if (nt + 1 < 32)
        __builtin_prefetch((const void*)(wbase + (n + 16) * 512), 0, 3);
      v8f acc = {};
#pragma unroll
      for (int kt = 0; kt < 16; ++kt) {
        bf16x16 bv = *(const bf16x16*)(wrow + (kt << 5));
        acc = wmma_bf16(ff[kt], bv, acc);
      }
      float bias = b_v[n];
      float p0 = 0.0f, p1 = 0.0f;
#pragma unroll
      for (int v = 0; v < 8; ++v) {
        float val = acc[v] + bias;
        p0 += e0[v] * val;
        p1 += e1[v] * val;
      }
      atomicAdd(&Bacc[n], p0);
      atomicAdd(&Bacc[512 + n], p1);
    }
  }
  __syncthreads();
  for (int i = tid; i < 1024; i += 256) atomicAdd(&BaccG[i], Bacc[i]);
}

__global__ void normA_kernel(float* Aout, const float* esum, int N) {
  int i = blockIdx.x * blockDim.x + threadIdx.x;
  if (i < 2 * N) Aout[i] = Aout[i] / esum[i & 1];
}

__global__ __launch_bounds__(256) void finalize_kernel(const float* BaccG, const float* esum,
                                                       const float* Wfcc, const float* b_fcc,
                                                       float* out, float* Bout) {
  __shared__ float red[256];
  float s0 = esum[0], s1 = esum[1];
  for (int i = threadIdx.x; i < 1024; i += 256)
    Bout[i] = BaccG[i] / ((i >> 9) ? s1 : s0);
  for (int o = 0; o < 2; ++o) {
    float part = 0.0f;
    for (int idx = threadIdx.x; idx < 1024; idx += 256) {
      int i = idx >> 9, d = idx & 511;
      part += (BaccG[idx] / (i ? s1 : s0)) * Wfcc[(o * 2 + i) * 512 + d];
    }
    red[threadIdx.x] = part;
    __syncthreads();
    for (int s = 128; s > 0; s >>= 1) {
      if (threadIdx.x < s) red[threadIdx.x] += red[threadIdx.x + s];
      __syncthreads();
    }
    if (threadIdx.x == 0) out[o] = red[0] + b_fcc[o];
    __syncthreads();
  }
}

extern "C" void kernel_launch(void* const* d_in, const int* in_sizes, int n_in,
                              void* d_out, int out_size, void* d_ws, size_t ws_size,
                              hipStream_t stream) {
  const float* feats = (const float*)d_in[0];
  const float* cmat  = (const float*)d_in[1];
  const float* preds = (const float*)d_in[2];
  const float* W_lin = (const float*)d_in[3];
  const float* b_lin = (const float*)d_in[4];
  const float* W_q   = (const float*)d_in[5];
  const float* b_q   = (const float*)d_in[6];
  const float* W_v   = (const float*)d_in[7];
  const float* b_v   = (const float*)d_in[8];
  const float* W_fcc = (const float*)d_in[9];
  const float* b_fcc = (const float*)d_in[10];

  int N = in_sizes[0] / 512;
  int k = (int)(0.2 * (double)N);
  if (k < 1) k = 1;

  float* out  = (float*)d_out;
  float* Aout = out + 2;
  float* Bout = out + 2 + (size_t)2 * N;

  char* ws = (char*)d_ws;
  unsigned* hist1 = (unsigned*)(ws + 0);
  unsigned* hist2 = (unsigned*)(ws + 8192);
  float*    scal  = (float*)(ws + 16384);
  unsigned* scalu = (unsigned*)(ws + 16384);
  unsigned long long* keys = (unsigned long long*)(ws + 16416);
  float* esum  = (float*)(ws + 16432);
  float* qmax  = (float*)(ws + 16448);
  float* BaccG = (float*)(ws + 17472);
  unsigned short* WlinB = (unsigned short*)(ws + 32768);
  unsigned short* WvB   = (unsigned short*)(ws + 557056);
  unsigned short* WqB   = (unsigned short*)(ws + 1081344);

  init_ws_kernel<<<32, 256, 0, stream>>>((unsigned*)ws, 5392);   // zero bytes [0,21568)
  conv_bf16_kernel<<<512, 256, 0, stream>>>(W_lin, W_v, W_q, WlinB, WvB, WqB);
  hist1_kernel<<<256, 256, 0, stream>>>(preds, hist1, N);
  scan1_kernel<<<1, 32, 0, stream>>>(hist1, scal, scalu, k);
  hist2_kernel<<<256, 256, 0, stream>>>(preds, hist2, scal, N);
  scan2_kernel<<<1, 32, 0, stream>>>(hist2, scal, scalu, k);
  maxred_kernel<<<256, 256, 0, stream>>>(preds, scal, scalu, N);
  expsum_kernel<<<256, 256, 0, stream>>>(preds, scal, scalu, N);
  argmaxc_kernel<<<128, 256, 0, stream>>>(cmat, keys, N);
  mfeat_kernel<<<2, 256, 0, stream>>>(feats, preds, W_lin, b_lin, W_q, b_q, keys, scal, scalu,
                                      qmax, N);

  int G = (N + TM - 1) / TM;
  const int SMEM = (2 * TM * XSTR) * 2 + (256 + 1024 + 2 + 128) * 4;  // 271,880 B < 320 KB
  main_kernel<<<G, 256, SMEM, stream>>>(feats, preds, WlinB, WvB, WqB, b_lin, b_v, b_q, scal,
                                        scalu, qmax, Aout, esum, BaccG, N);
  normA_kernel<<<(2 * N + 255) / 256, 256, 0, stream>>>(Aout, esum, N);
  finalize_kernel<<<1, 256, 0, stream>>>(BaccG, esum, W_fcc, b_fcc, out, Bout);
}
